// DecoderLayer_75514114998542
// MI455X (gfx1250) — matrix operand order repaired and checked by reading, not compiled, not run on hardware
//
#include <hip/hip_runtime.h>

// ---------------------------------------------------------------------------
// Decoder layer for MI455X (gfx1250, wave32, WMMA).
// bf16 WMMA (v_wmma_f32_16x16x32_bf16) everywhere; fp32 accumulate.
// Projection GEMMs stage weights via global_load_async_to_lds_b128 (ASYNCcnt,
// double-buffered, 64-K stages = 2 WMMAs per barrier window).
// Sparse top-8 grouped-GEMM MoE via per-expert token lists.
// ---------------------------------------------------------------------------

typedef __attribute__((ext_vector_type(16))) __bf16 v16bf;
typedef __attribute__((ext_vector_type(8)))  float  v8f;

#define DEVFN __device__ __forceinline__

union Frag {
  v16bf v;
  uint4 q[2];
  unsigned u[8];
  unsigned short s[16];
};

DEVFN unsigned fu(float x) { union { float f; unsigned u; } v; v.f = x; return v.u; }

DEVFN unsigned short f2bf(float f) {
  unsigned u = fu(f);
  return (unsigned short)((u + 0x7FFFu + ((u >> 16) & 1u)) >> 16);
}

// pack two f32 -> two bf16 (truncate) in one v_perm_b32
DEVFN unsigned pk_bf16(float lo, float hi) {
  return __builtin_amdgcn_perm(fu(hi), fu(lo), 0x07060302u);
}

// convert 16 consecutive f32 -> bf16 fragment half-lane chunk (8 v_perm)
DEVFN void conv16(const float* __restrict__ src, Frag& b) {
  const float4* s4 = (const float4*)src;
  #pragma unroll
  for (int i = 0; i < 4; i++) {
    float4 f = s4[i];
    b.u[2 * i]     = pk_bf16(f.x, f.y);
    b.u[2 * i + 1] = pk_bf16(f.z, f.w);
  }
}

DEVFN v8f wmma_bf16(const Frag& a, const Frag& b, v8f c) {
  return __builtin_amdgcn_wmma_f32_16x16x32_bf16(false, a.v, false, b.v,
                                                 (short)0, c, false, false);
}

DEVFN Frag load_b16(const unsigned short* p) {
  Frag f;
  f.q[0] = *(const uint4*)(p);
  f.q[1] = *(const uint4*)(p + 8);
  return f;
}

DEVFN float redmax16(float v) {
  #pragma unroll
  for (int m = 1; m < 16; m <<= 1) v = fmaxf(v, __shfl_xor(v, m, 16));
  return v;
}
DEVFN float redsum16(float v) {
  #pragma unroll
  for (int m = 1; m < 16; m <<= 1) v += __shfl_xor(v, m, 16);
  return v;
}

// LDS byte offset of a __shared__ object (addrspace(3) pointers are 32-bit)
DEVFN unsigned lds_off(const void* p) {
  return (unsigned)(unsigned long long)(__attribute__((address_space(3))) const void*)p;
}

// async copy 16B global -> LDS (per-lane addresses), tracked by ASYNCcnt
DEVFN void async_b128(unsigned ldsoff, const void* gaddr) {
  asm volatile("global_load_async_to_lds_b128 %0, %1, off"
               :: "v"(ldsoff), "v"(gaddr) : "memory");
}
#define WAIT_ASYNC(n) asm volatile("s_wait_asynccnt %0" :: "n"(n) : "memory")

// ----- problem constants ----------------------------------------------------
constexpr int Bt = 2, S = 1024, D = 2048;
constexpr int H = 32, HKV = 4, HD = 128;
constexpr int E = 64, TOPK = 8, MI = 768;
constexpr int T = Bt * S;
constexpr int QD = H * HD;           // 4096
constexpr int KVD = HKV * HD;        // 512
constexpr int TOT_ROWS = T * TOPK;   // 16384

// ----- workspace layout (bytes) --------------------------------------------
constexpr size_t XN_OFF  = 0;
constexpr size_t QF_OFF  = XN_OFF  + (size_t)T * D * 2;
constexpr size_t KF_OFF  = QF_OFF  + (size_t)T * QD * 4;
constexpr size_t VF_OFF  = KF_OFF  + (size_t)T * KVD * 4;
constexpr size_t QB_OFF  = VF_OFF  + (size_t)T * KVD * 4;
constexpr size_t KB_OFF  = QB_OFF  + (size_t)T * QD * 2;
constexpr size_t VB_OFF  = KB_OFF  + (size_t)T * KVD * 2;
constexpr size_t AO_OFF  = VB_OFF  + (size_t)T * KVD * 2;
constexpr size_t OP_OFF  = AO_OFF  + (size_t)T * QD * 2;
constexpr size_t XN2_OFF = OP_OFF  + (size_t)T * D * 4;
constexpr size_t LG_OFF  = XN2_OFF + (size_t)T * D * 2;
constexpr size_t CB_OFF  = LG_OFF  + (size_t)T * E * 4;
constexpr size_t CNT_OFF = CB_OFF  + (size_t)T * E * 4;
constexpr size_t OFS_OFF = CNT_OFF + 512;
constexpr size_t LST_OFF = OFS_OFF + 512;
constexpr size_t GU_OFF  = LST_OFF + (size_t)E * T * 4;
constexpr size_t HB_OFF  = GU_OFF  + (size_t)TOT_ROWS * (2 * MI) * 4;

// ---------------------------------------------------------------------------
__global__ void rmsnorm_bf16_kernel(const float* __restrict__ x,
                                    const float* __restrict__ w,
                                    unsigned short* __restrict__ out, int cols) {
  __shared__ float red[256];
  int row = blockIdx.x;
  const float* xr = x + (size_t)row * cols;
  float acc = 0.f;
  for (int c = threadIdx.x; c < cols; c += blockDim.x) { float v = xr[c]; acc += v * v; }
  red[threadIdx.x] = acc; __syncthreads();
  for (int s = blockDim.x >> 1; s > 0; s >>= 1) {
    if ((int)threadIdx.x < s) red[threadIdx.x] += red[threadIdx.x + s];
    __syncthreads();
  }
  float inv = rsqrtf(red[0] / (float)cols + 1e-6f);
  unsigned short* o = out + (size_t)row * cols;
  for (int c = threadIdx.x; c < cols; c += blockDim.x) o[c] = f2bf(w[c] * xr[c] * inv);
}

__global__ void headnorm_bf16_kernel(const float* __restrict__ x,
                                     const float* __restrict__ w,
                                     unsigned short* __restrict__ out, float scale) {
  __shared__ float red[128];
  size_t row = blockIdx.x;
  float v = x[row * HD + threadIdx.x];
  red[threadIdx.x] = v * v; __syncthreads();
  for (int s = 64; s > 0; s >>= 1) {
    if ((int)threadIdx.x < s) red[threadIdx.x] += red[threadIdx.x + s];
    __syncthreads();
  }
  float inv = rsqrtf(red[0] / (float)HD + 1e-6f);
  out[row * HD + threadIdx.x] = f2bf(w[threadIdx.x] * v * inv * scale);
}

__global__ void f32_to_bf16_kernel(const float* __restrict__ in,
                                   unsigned short* __restrict__ out, int n) {
  int i = blockIdx.x * blockDim.x + threadIdx.x;
  if (i < n) out[i] = f2bf(in[i]);
}

// ---------------------------------------------------------------------------
// C[M,N] = A(bf16 MxK) * W(f32 NxK)^T.  block=256 (8 waves, 2x4), tile 32x64.
// Weight stage: 64 cols x 64 K fp32 (16KB) double-buffered via
// global_load_async_to_lds_b128; 2 WMMAs per barrier window.
// All launch dims are exact tile multiples -> no bounds checks.
// ---------------------------------------------------------------------------
__global__ void gemm_bf16_kernel(const unsigned short* __restrict__ A,
                                 const float* __restrict__ W,
                                 float* __restrict__ C, int M, int N, int K) {
  __shared__ __align__(16) float Bs[2][64 * 64];   // [buf][col*64 + k]
  int t = threadIdx.x;
  int lane = t & 31, wave = t >> 5;
  int wm = wave >> 2, wn = wave & 3;
  int row0 = (blockIdx.x * 2 + wm) * 16;
  int colbase = blockIdx.y * 64;
  int half = lane >> 4, mr = lane & 15;

  // staging: 1024 16B-chunks per 64x64 tile, 4 per thread.
  // chunk c: col = c>>4, koff = (c&15)*4 floats.
  const float* g[4];
  unsigned rel[4];
  #pragma unroll
  for (int i = 0; i < 4; i++) {
    int c = t + 256 * i;
    g[i] = W + (size_t)(colbase + (c >> 4)) * K + (c & 15) * 4;
    rel[i] = (unsigned)((c >> 4) * 256 + (c & 15) * 16);   // bytes
  }
  unsigned lb0 = lds_off(&Bs[0][0]);
  unsigned lb1 = lds_off(&Bs[1][0]);

  #pragma unroll
  for (int i = 0; i < 4; i++) async_b128(lb0 + rel[i], g[i]);

  const unsigned short* Ap = A + (size_t)(row0 + mr) * K;
  const float* bs0 = &Bs[0][(wn * 16 + mr) * 64];
  const float* bs1 = &Bs[1][(wn * 16 + mr) * 64];
  v8f acc = {0.f, 0.f, 0.f, 0.f, 0.f, 0.f, 0.f, 0.f};
  int nst = K >> 6;
  for (int st = 0; st < nst; st++) {
    int cur = st & 1;
    if (st + 1 < nst) {
      unsigned lb = cur ? lb0 : lb1;
      #pragma unroll
      for (int i = 0; i < 4; i++) async_b128(lb + rel[i], g[i] + (size_t)(st + 1) * 64);
      WAIT_ASYNC(4);
    } else {
      WAIT_ASYNC(0);
    }
    __syncthreads();
    const float* bsc = cur ? bs1 : bs0;
    #pragma unroll
    for (int kk = 0; kk < 2; kk++) {
      int k0 = (st << 6) + (kk << 5);
      Frag a, b;
      a.q[0] = *(const uint4*)(Ap + k0 + half * 16);
      a.q[1] = *(const uint4*)(Ap + k0 + half * 16 + 8);
      conv16(bsc + (kk << 5) + half * 16, b);
      acc = wmma_bf16(a, b, acc);
    }
    __syncthreads();
  }
  float* Cp = C + (size_t)(row0 + 8 * half) * N + colbase + wn * 16 + mr;
  #pragma unroll
  for (int r = 0; r < 8; r++) Cp[(size_t)r * N] = acc[r];
}

// ---------------------------------------------------------------------------
// FlashAttention-style causal attention, one wave per (b, h, 16 q-rows).
// ---------------------------------------------------------------------------
__global__ void attn_kernel(const unsigned short* __restrict__ Qb,
                            const unsigned short* __restrict__ Kb,
                            const unsigned short* __restrict__ Vb,
                            unsigned short* __restrict__ Ob) {
  __shared__ __align__(16) unsigned short Vt[HD * 32];
  __shared__ __align__(16) unsigned short Plds[16 * 32];
  int q0 = blockIdx.x * 16, h = blockIdx.y, b = blockIdx.z;
  int kvh = h >> 3;
  int lane = threadIdx.x & 31, half = lane >> 4, mr = lane & 15;

  Frag qa[4];
  {
    const unsigned short* qp = Qb + (size_t)(b * S + q0 + mr) * QD + h * HD;
    #pragma unroll
    for (int kk = 0; kk < 4; kk++) {
      qa[kk].q[0] = *(const uint4*)(qp + kk * 32 + half * 16);
      qa[kk].q[1] = *(const uint4*)(qp + kk * 32 + half * 16 + 8);
    }
  }

  v8f o[8];
  float rowmax[8], rowsum[8];
  #pragma unroll
  for (int j = 0; j < 8; j++) o[j] = (v8f){0.f,0.f,0.f,0.f,0.f,0.f,0.f,0.f};
  #pragma unroll
  for (int r = 0; r < 8; r++) { rowmax[r] = -1e30f; rowsum[r] = 0.f; }

  for (int j0 = 0; j0 <= q0 + 15; j0 += 32) {
    {
      int kpos = j0 + lane; int kposc = kpos < S ? kpos : S - 1;
      const unsigned short* vp = Vb + (size_t)(b * S + kposc) * KVD + kvh * HD;
      #pragma unroll
      for (int c = 0; c < HD; c += 8) {
        uint4 dv = *(const uint4*)(vp + c);
        const unsigned short* dvp = (const unsigned short*)&dv;
        #pragma unroll
        for (int i = 0; i < 8; i++) Vt[(c + i) * 32 + lane] = dvp[i];
      }
    }
    v8f s0 = (v8f){0.f,0.f,0.f,0.f,0.f,0.f,0.f,0.f};
    v8f s1 = s0;
    {
      int kp0 = j0 + mr;      int kp0c = kp0 < S ? kp0 : S - 1;
      int kp1 = j0 + 16 + mr; int kp1c = kp1 < S ? kp1 : S - 1;
      const unsigned short* k0p = Kb + (size_t)(b * S + kp0c) * KVD + kvh * HD;
      const unsigned short* k1p = Kb + (size_t)(b * S + kp1c) * KVD + kvh * HD;
      #pragma unroll
      for (int kk = 0; kk < 4; kk++) {
        Frag kb0 = load_b16(k0p + kk * 32 + half * 16);
        s0 = wmma_bf16(qa[kk], kb0, s0);
        Frag kb1 = load_b16(k1p + kk * 32 + half * 16);
        s1 = wmma_bf16(qa[kk], kb1, s1);
      }
    }
    float ps0[8], ps1[8];
    #pragma unroll
    for (int r = 0; r < 8; r++) {
      int qrow = q0 + r + 8 * half;
      float v0 = ((j0 + mr) <= qrow) ? s0[r] : -1e30f;
      float v1 = ((j0 + 16 + mr) <= qrow) ? s1[r] : -1e30f;
      float mnew = redmax16(fmaxf(v0, v1));
      float m = fmaxf(rowmax[r], mnew);
      float alpha = __expf(rowmax[r] - m);
      rowmax[r] = m;
      float p0 = __expf(v0 - m), p1 = __expf(v1 - m);
      ps0[r] = p0; ps1[r] = p1;
      rowsum[r] = rowsum[r] * alpha + redsum16(p0 + p1);
      #pragma unroll
      for (int j = 0; j < 8; j++) o[j][r] *= alpha;
    }
    #pragma unroll
    for (int r = 0; r < 8; r++) {
      int row = r + 8 * half;
      Plds[row * 32 + mr]      = f2bf(ps0[r]);
      Plds[row * 32 + 16 + mr] = f2bf(ps1[r]);
    }
    __syncthreads();
    Frag pa;
    pa.q[0] = *(const uint4*)&Plds[mr * 32 + half * 16];
    pa.q[1] = *(const uint4*)&Plds[mr * 32 + half * 16 + 8];
    #pragma unroll
    for (int j = 0; j < 8; j++) {
      Frag vb;
      vb.q[0] = *(const uint4*)&Vt[(j * 16 + mr) * 32 + half * 16];
      vb.q[1] = *(const uint4*)&Vt[(j * 16 + mr) * 32 + half * 16 + 8];
      o[j] = wmma_bf16(pa, vb, o[j]);
    }
    __syncthreads();
  }
  #pragma unroll
  for (int r = 0; r < 8; r++) {
    int row = q0 + r + 8 * half;
    float invl = 1.0f / rowsum[r];
    unsigned short* op = Ob + (size_t)(b * S + row) * QD + h * HD;
    #pragma unroll
    for (int j = 0; j < 8; j++) op[j * 16 + mr] = f2bf(o[j][r] * invl);
  }
}

__global__ void residual_add_kernel(const float* __restrict__ a,
                                    const float* __restrict__ b,
                                    float* __restrict__ out, int n) {
  int i = blockIdx.x * blockDim.x + threadIdx.x;
  if (i < n) out[i] = a[i] + b[i];
}

__global__ void zero_kernel(float* __restrict__ cb, int n,
                            int* __restrict__ cnts, int nc) {
  int i = blockIdx.x * blockDim.x + threadIdx.x;
  if (i < n) cb[i] = 0.f;
  if (i < nc) cnts[i] = 0;
}

__global__ void router_topk_kernel(const float* __restrict__ logits,
                                   float* __restrict__ combine,
                                   int* __restrict__ counts,
                                   int* __restrict__ lists) {
  int t = blockIdx.x * blockDim.x + threadIdx.x;
  if (t >= T) return;
  float p[E];
  float mx = -1e30f;
  for (int e = 0; e < E; e++) { p[e] = logits[(size_t)t * E + e]; mx = fmaxf(mx, p[e]); }
  float sum = 0.f;
  for (int e = 0; e < E; e++) { p[e] = __expf(p[e] - mx); sum += p[e]; }
  float invs = 1.f / sum;
  for (int e = 0; e < E; e++) p[e] *= invs;
  int idxs[TOPK]; float vals[TOPK]; float vsum = 0.f;
  for (int k = 0; k < TOPK; k++) {
    int best = 0; float bv = -1.f;
    for (int e = 0; e < E; e++) if (p[e] > bv) { bv = p[e]; best = e; }
    idxs[k] = best; vals[k] = bv; vsum += bv; p[best] = -1.f;
  }
  float invv = 1.f / vsum;
  for (int k = 0; k < TOPK; k++) {
    combine[(size_t)t * E + idxs[k]] = vals[k] * invv;
    int pos = atomicAdd(&counts[idxs[k]], 1);
    lists[(size_t)idxs[k] * T + pos] = t;
  }
}

__global__ void scan_kernel(const int* __restrict__ counts, int* __restrict__ offs) {
  if (threadIdx.x == 0) {
    int a = 0;
    for (int e = 0; e < E; e++) { offs[e] = a; a += counts[e]; }
    offs[E] = a;
  }
}

// ---------------------------------------------------------------------------
// MoE gate_up grouped GEMM (gathered rows). grid (T/16, 2MI/64, E), block 128.
// ---------------------------------------------------------------------------
__global__ void moe_gateup_kernel(const unsigned short* __restrict__ Xb,
                                  const float* __restrict__ GU,
                                  const int* __restrict__ counts,
                                  const int* __restrict__ offs,
                                  const int* __restrict__ lists,
                                  float* __restrict__ gubuf) {
  int e = blockIdx.z;
  int cnt = counts[e];
  int rt = blockIdx.x;
  if (rt * 16 >= cnt) return;
  int lane = threadIdx.x & 31, wave = threadIdx.x >> 5;
  int half = lane >> 4, mr = lane & 15;
  int col0 = blockIdx.y * 64 + wave * 16;
  int ridx = rt * 16 + mr; if (ridx >= cnt) ridx = cnt - 1;
  int tok = lists[(size_t)e * T + ridx];
  const unsigned short* Ap = Xb + (size_t)tok * D;
  const float* Wp = GU + ((size_t)e * (2 * MI) + col0 + mr) * D;
  v8f acc = {0.f,0.f,0.f,0.f,0.f,0.f,0.f,0.f};
  #pragma unroll 2
  for (int k0 = 0; k0 < D; k0 += 32) {
    Frag a, b;
    a.q[0] = *(const uint4*)(Ap + k0 + half * 16);
    a.q[1] = *(const uint4*)(Ap + k0 + half * 16 + 8);
    conv16(Wp + k0 + half * 16, b);
    __builtin_prefetch(Wp + k0 + 128, 0, 1);
    acc = wmma_bf16(a, b, acc);
  }
  int base = offs[e];
  #pragma unroll
  for (int r = 0; r < 8; r++) {
    int rr = rt * 16 + r + 8 * half;
    if (rr < cnt) gubuf[(size_t)(base + rr) * (2 * MI) + col0 + mr] = acc[r];
  }
}

__global__ void silu_mul_kernel(const float* __restrict__ gu,
                                unsigned short* __restrict__ hb) {
  size_t i = (size_t)blockIdx.x * blockDim.x + threadIdx.x;
  size_t row = i / MI, j = i % MI;
  float g = gu[row * (2 * MI) + j];
  float u = gu[row * (2 * MI) + MI + j];
  float s = g / (1.f + __expf(-g));
  hb[i] = f2bf(s * u);
}

// ---------------------------------------------------------------------------
// MoE down grouped GEMM + weighted scatter-add. grid (T/16, D/64, E), block 128.
// ---------------------------------------------------------------------------
__global__ void moe_down_kernel(const unsigned short* __restrict__ Hb,
                                const float* __restrict__ DW,
                                const int* __restrict__ counts,
                                const int* __restrict__ offs,
                                const int* __restrict__ lists,
                                const float* __restrict__ combine,
                                float* __restrict__ out) {
  int e = blockIdx.z;
  int cnt = counts[e];
  int rt = blockIdx.x;
  if (rt * 16 >= cnt) return;
  int lane = threadIdx.x & 31, wave = threadIdx.x >> 5;
  int half = lane >> 4, mr = lane & 15;
  int col0 = blockIdx.y * 64 + wave * 16;
  int ridx = rt * 16 + mr; if (ridx >= cnt) ridx = cnt - 1;
  const unsigned short* Ap = Hb + (size_t)(offs[e] + ridx) * MI;
  const float* Wp = DW + ((size_t)e * D + col0 + mr) * MI;
  v8f acc = {0.f,0.f,0.f,0.f,0.f,0.f,0.f,0.f};
  #pragma unroll 2
  for (int k0 = 0; k0 < MI; k0 += 32) {
    Frag a, b;
    a.q[0] = *(const uint4*)(Ap + k0 + half * 16);
    a.q[1] = *(const uint4*)(Ap + k0 + half * 16 + 8);
    conv16(Wp + k0 + half * 16, b);
    __builtin_prefetch(Wp + k0 + 128, 0, 1);
    acc = wmma_bf16(a, b, acc);
  }
  #pragma unroll
  for (int r = 0; r < 8; r++) {
    int rr = rt * 16 + r + 8 * half;
    if (rr < cnt) {
      int tok = lists[(size_t)e * T + rr];
      float cw = combine[(size_t)tok * E + e];
      atomicAdd(out + (size_t)tok * D + col0 + mr, cw * acc[r]);
    }
  }
}

// ---------------------------------------------------------------------------
extern "C" void kernel_launch(void* const* d_in, const int* in_sizes, int n_in,
                              void* d_out, int out_size, void* d_ws, size_t ws_size,
                              hipStream_t stream) {
  (void)in_sizes; (void)n_in; (void)out_size; (void)ws_size;
  const float* x         = (const float*)d_in[0];
  const float* in_ln_w   = (const float*)d_in[1];
  const float* q_w       = (const float*)d_in[2];
  const float* k_w       = (const float*)d_in[3];
  const float* v_w       = (const float*)d_in[4];
  const float* o_w       = (const float*)d_in[5];
  const float* qn_w      = (const float*)d_in[6];
  const float* kn_w      = (const float*)d_in[7];
  const float* post_ln_w = (const float*)d_in[8];
  const float* router_w  = (const float*)d_in[9];
  const float* gate_up_w = (const float*)d_in[10];
  const float* down_w    = (const float*)d_in[11];
  float* out = (float*)d_out;

  char* w = (char*)d_ws;
  unsigned short* xnb  = (unsigned short*)(w + XN_OFF);
  float*          qf   = (float*)(w + QF_OFF);
  float*          kf   = (float*)(w + KF_OFF);
  float*          vf   = (float*)(w + VF_OFF);
  unsigned short* qb   = (unsigned short*)(w + QB_OFF);
  unsigned short* kb   = (unsigned short*)(w + KB_OFF);
  unsigned short* vb   = (unsigned short*)(w + VB_OFF);
  unsigned short* aob  = (unsigned short*)(w + AO_OFF);
  float*          opf  = (float*)(w + OP_OFF);
  unsigned short* xn2b = (unsigned short*)(w + XN2_OFF);
  float*          lg   = (float*)(w + LG_OFF);
  float*          cbw  = (float*)(w + CB_OFF);
  int*            cnts = (int*)(w + CNT_OFF);
  int*            ofs  = (int*)(w + OFS_OFF);
  int*            lst  = (int*)(w + LST_OFF);
  float*          gub  = (float*)(w + GU_OFF);
  unsigned short* hb   = (unsigned short*)(w + HB_OFF);

  const float qscale = 0.08838834764831845f; // HD^-0.5

  rmsnorm_bf16_kernel<<<T, 256, 0, stream>>>(x, in_ln_w, xnb, D);
  gemm_bf16_kernel<<<dim3(T/32, QD/64), 256, 0, stream>>>(xnb, q_w, qf, T, QD, D);
  gemm_bf16_kernel<<<dim3(T/32, KVD/64), 256, 0, stream>>>(xnb, k_w, kf, T, KVD, D);
  gemm_bf16_kernel<<<dim3(T/32, KVD/64), 256, 0, stream>>>(xnb, v_w, vf, T, KVD, D);
  headnorm_bf16_kernel<<<T*H, 128, 0, stream>>>(qf, qn_w, qb, qscale);
  headnorm_bf16_kernel<<<T*HKV, 128, 0, stream>>>(kf, kn_w, kb, 1.0f);
  f32_to_bf16_kernel<<<(T*KVD)/256, 256, 0, stream>>>(vf, vb, T*KVD);
  attn_kernel<<<dim3(S/16, H, Bt), 32, 0, stream>>>(qb, kb, vb, aob);
  gemm_bf16_kernel<<<dim3(T/32, D/64), 256, 0, stream>>>(aob, o_w, opf, T, D, QD);
  residual_add_kernel<<<(T*D)/256, 256, 0, stream>>>(x, opf, out, T*D);
  rmsnorm_bf16_kernel<<<T, 256, 0, stream>>>(out, post_ln_w, xn2b, D);
  gemm_bf16_kernel<<<dim3(T/32, 1), 256, 0, stream>>>(xn2b, router_w, lg, T, E, D);
  zero_kernel<<<(T*E)/256, 256, 0, stream>>>(cbw, T*E, cnts, E);
  router_topk_kernel<<<T/256, 256, 0, stream>>>(lg, cbw, cnts, lst);
  scan_kernel<<<1, 32, 0, stream>>>(cnts, ofs);
  moe_gateup_kernel<<<dim3(T/16, (2*MI)/64, E), 128, 0, stream>>>(
      xn2b, gate_up_w, cnts, ofs, lst, gub);
  silu_mul_kernel<<<((size_t)TOT_ROWS * MI) / 256, 256, 0, stream>>>(gub, hb);
  moe_down_kernel<<<dim3(T/16, D/64, E), 128, 0, stream>>>(
      hb, down_w, cnts, ofs, lst, cbw, out);
}
